// TurboAGaLiTeLayer_20031727468689
// MI455X (gfx1250) — matrix operand order repaired and checked by reading, not compiled, not run on hardware
//
#include <hip/hip_runtime.h>

// ---------------------------------------------------------------------------
// Types for CDNA5 WMMA (wave32)
// ---------------------------------------------------------------------------
typedef __bf16 bf16_t;
typedef bf16_t bf16x16 __attribute__((ext_vector_type(16)));
typedef float  floatx8 __attribute__((ext_vector_type(8)));

#define TT   1024
#define BB   8
#define DD   512
#define HH   8
#define DHH  64
#define ETA_ 2
#define RR   4
#define DKE_ 128
#define DFFC_ 2048
#define NPROJ 2608   // H*DH*5 + H*ETA*3
#define KQVD  2560   // H*DH*5

__device__ __forceinline__ float sigmoidf_(float x) {
    return 1.0f / (1.0f + __expf(-x));
}

// A fragment: 16x32 bf16, row-major source with leading dim ldk.
// lane: m = lane%16, half = lane/16.
// elems 0..7  <- K = kb + half*8 .. +7
// elems 8..15 <- K = kb + 16 + half*8 .. +7
__device__ __forceinline__ bf16x16 load_frag_a(const bf16_t* __restrict__ base,
                                               int row, int ldk, int kb, int half_) {
    const bf16_t* p0 = base + (size_t)row * ldk + kb + half_ * 8;
    union { uint4 u[2]; bf16x16 v; } u;
    u.u[0] = *reinterpret_cast<const uint4*>(p0);
    u.u[1] = *reinterpret_cast<const uint4*>(p0 + 16);
    return u.v;
}

// B fragment: 32x16 bf16 from W stored (N,K) row-major (i.e. W^T GEMM operand).
// lane: n = lane%16, half = lane/16.
// elems 0..7  <- K = kb + half*8 .. +7
// elems 8..15 <- K = kb + 16 + half*8 .. +7
__device__ __forceinline__ bf16x16 load_frag_b(const bf16_t* __restrict__ base,
                                               int row, int ldk, int kb, int half_) {
    const bf16_t* p0 = base + (size_t)row * ldk + kb + half_ * 8;
    union { uint4 u[2]; bf16x16 v; } u;
    u.u[0] = *reinterpret_cast<const uint4*>(p0);
    u.u[1] = *reinterpret_cast<const uint4*>(p0 + 16);
    return u.v;
}

// ---------------------------------------------------------------------------
// Generic bf16 WMMA GEMM: out[m,n] = act( A[m,:] . W[n,:] + bias[n] + res[m,n] )
// A: (M,K) bf16 row-major, W: (N,K) bf16 row-major. M % 64 == 0, K % 32 == 0.
// Block: 128 threads (4 waves, 2x2), block tile 64x64, wave tile 32x32.
// ---------------------------------------------------------------------------
__global__ __launch_bounds__(128)
void gemm_bf16_wmma(const bf16_t* __restrict__ A, const bf16_t* __restrict__ W,
                    const float* __restrict__ bias, const float* __restrict__ residual,
                    float* __restrict__ outF, bf16_t* __restrict__ outH,
                    int M, int N, int K, int applyRelu) {
    const int wave  = threadIdx.x >> 5;
    const int lane  = threadIdx.x & 31;
    const int half_ = lane >> 4;
    const int lr    = lane & 15;
    const int waveM = wave >> 1;
    const int waveN = wave & 1;

    const int m0 = blockIdx.y * 64 + waveM * 32;
    const int n0 = blockIdx.x * 64 + waveN * 32;

    floatx8 acc[2][2] = {};

    const int rowA0 = m0 + lr;
    const int rowA1 = m0 + 16 + lr;
    int nB0 = n0 + lr;        if (nB0 >= N) nB0 = N - 1;   // clamp loads, mask stores
    int nB1 = n0 + 16 + lr;   if (nB1 >= N) nB1 = N - 1;

    for (int kb = 0; kb < K; kb += 32) {
        bf16x16 a0 = load_frag_a(A, rowA0, K, kb, half_);
        bf16x16 a1 = load_frag_a(A, rowA1, K, kb, half_);
        bf16x16 b0 = load_frag_b(W, nB0, K, kb, half_);
        bf16x16 b1 = load_frag_b(W, nB1, K, kb, half_);
        acc[0][0] = __builtin_amdgcn_wmma_f32_16x16x32_bf16(false, a0, false, b0, (short)0, acc[0][0], false, false);
        acc[0][1] = __builtin_amdgcn_wmma_f32_16x16x32_bf16(false, a0, false, b1, (short)0, acc[0][1], false, false);
        acc[1][0] = __builtin_amdgcn_wmma_f32_16x16x32_bf16(false, a1, false, b0, (short)0, acc[1][0], false, false);
        acc[1][1] = __builtin_amdgcn_wmma_f32_16x16x32_bf16(false, a1, false, b1, (short)0, acc[1][1], false, false);
    }

    // D layout: lane n = lane%16, m = v + half*8  (per 16x16 tile)
    for (int mi = 0; mi < 2; ++mi) {
        for (int ni = 0; ni < 2; ++ni) {
            const int n = n0 + ni * 16 + lr;
            if (n >= N) continue;
            const int mbase = m0 + mi * 16 + half_ * 8;
            const float bv = bias ? bias[n] : 0.0f;
            #pragma unroll
            for (int v = 0; v < 8; ++v) {
                const int m = mbase + v;
                float val = acc[mi][ni][v] + bv;
                if (residual) val += residual[(size_t)m * N + n];
                if (applyRelu) val = fmaxf(val, 0.0f);
                if (outF) outF[(size_t)m * N + n] = val;
                if (outH) outH[(size_t)m * N + n] = (bf16_t)val;
            }
        }
    }
}

// ---------------------------------------------------------------------------
// LayerNorm over rows of (rows, D); writes f32 and/or bf16.
// ---------------------------------------------------------------------------
__global__ __launch_bounds__(256)
void ln_kernel(const float* __restrict__ in, const float* __restrict__ g,
               const float* __restrict__ bta, float* __restrict__ outF,
               bf16_t* __restrict__ outH, int D) {
    const int row = blockIdx.x;
    const float* x = in + (size_t)row * D;
    float s = 0.0f, s2 = 0.0f;
    for (int i = threadIdx.x; i < D; i += 256) { float v = x[i]; s += v; s2 += v * v; }
    #pragma unroll
    for (int off = 16; off; off >>= 1) {
        s  += __shfl_down(s,  off, 32);
        s2 += __shfl_down(s2, off, 32);
    }
    __shared__ float ps[8], ps2[8];
    __shared__ float mu_s, rstd_s;
    const int wv = threadIdx.x >> 5, ln = threadIdx.x & 31;
    if (ln == 0) { ps[wv] = s; ps2[wv] = s2; }
    __syncthreads();
    if (threadIdx.x == 0) {
        float ts = 0.0f, ts2 = 0.0f;
        #pragma unroll
        for (int i = 0; i < 8; ++i) { ts += ps[i]; ts2 += ps2[i]; }
        const float mu  = ts / (float)D;
        const float var = ts2 / (float)D - mu * mu;
        mu_s = mu;
        rstd_s = rsqrtf(var + 1e-5f);
    }
    __syncthreads();
    const float mu = mu_s, rs = rstd_s;
    for (int i = threadIdx.x; i < D; i += 256) {
        const float v = (x[i] - mu) * rs * g[i] + bta[i];
        if (outF) outF[(size_t)row * D + i] = v;
        if (outH) outH[(size_t)row * D + i] = (bf16_t)v;
    }
}

// ---------------------------------------------------------------------------
// f32 -> bf16 cast (grid-stride)
// ---------------------------------------------------------------------------
__global__ void cast_f32_bf16(const float* __restrict__ in, bf16_t* __restrict__ out, int n) {
    for (int i = blockIdx.x * blockDim.x + threadIdx.x; i < n; i += gridDim.x * blockDim.x)
        out[i] = (bf16_t)in[i];
}

// ---------------------------------------------------------------------------
// Fused gating + oscillator + discounted scans + attention readout.
// One block per (b,h) channel, 256 threads (8 waves). State lives in LDS.
//   s_tk[r][j], s_tv[r][d], s_s[j] carried across the T=1024 loop.
// Writes attn (T*B, H*DH) as bf16 for the following WMMA GEMM.
// ---------------------------------------------------------------------------
__global__ __launch_bounds__(256)
void scan_attn_kernel(const float* __restrict__ all_proj,    // (T*B, 2608)
                      const float* __restrict__ terminations,// (T, B)
                      const float* __restrict__ tk_prev,     // (B, R, H, DKE)
                      const float* __restrict__ tv_prev,     // (B, R, H, DH)
                      const float* __restrict__ s_prev,      // (B, H, DKE)
                      const float* __restrict__ tick,        // (B, 1)
                      const float* __restrict__ omegas,      // (R)
                      bf16_t* __restrict__ attn_bf)          // (T*B, 512)
{
    const int b = blockIdx.x >> 3;
    const int h = blockIdx.x & 7;
    const int tid = threadIdx.x;

    __shared__ float s_tk[RR][DKE_];
    __shared__ float s_tv[RR][DHH];
    __shared__ float s_s[DKE_];
    __shared__ float s_q[DKE_];
    __shared__ float s_red[8];   // dot[0..3] at 0..3, den at 4

    if (tid < DKE_) {
        s_s[tid] = s_prev[((size_t)b * HH + h) * DKE_ + tid];
        #pragma unroll
        for (int r = 0; r < RR; ++r)
            s_tk[r][tid] = tk_prev[(((size_t)b * RR + r) * HH + h) * DKE_ + tid];
    } else if (tid < DKE_ + DHH) {
        const int d = tid - DKE_;
        #pragma unroll
        for (int r = 0; r < RR; ++r)
            s_tv[r][d] = tv_prev[(((size_t)b * RR + r) * HH + h) * DHH + d];
    }
    const float tick0 = tick[b];
    float om[RR];
    #pragma unroll
    for (int r = 0; r < RR; ++r) om[r] = omegas[r];
    __syncthreads();

    for (int t = 0; t < TT; ++t) {
        const float* ap = all_proj + ((size_t)t * BB + b) * NPROJ;
        const float tm = 1.0f - terminations[t * BB + b];
        const float tkv = tick0 + (float)(t + 1);
        float occ[RR];
        #pragma unroll
        for (int r = 0; r < RR; ++r) occ[r] = __cosf(tkv * om[r]);

        if (tid < DKE_) {
            const int d = tid >> 1, e = tid & 1;
            const float* kq = ap + h * (5 * DHH);
            const float key = kq[d];
            const float qv  = kq[1 * DHH + d];
            const float gam = kq[4 * DHH + d];
            const float* pp = ap + KQVD + h * (3 * ETA_);
            const float p1 = pp[e], p2 = pp[ETA_ + e], p3 = pp[2 * ETA_ + e];
            const float kg  = fmaxf(key, 0.0f) * sigmoidf_(gam);
            const float pg  = fmaxf(p1, 0.0f)  * sigmoidf_(p3);
            const float kgp = kg * pg;                         // keys_gated[j]
            s_q[tid] = fmaxf(qv, 0.0f) * fmaxf(p2, 0.0f);      // queries_expanded[j]
            const float dg = (1.0f - kgp) * tm;
            s_s[tid] = dg * s_s[tid] + kgp;
            #pragma unroll
            for (int r = 0; r < RR; ++r)
                s_tk[r][tid] = dg * s_tk[r][tid] + kgp * occ[r];
        } else if (tid < DKE_ + DHH) {
            const int d = tid - DKE_;
            const float* kq = ap + h * (5 * DHH);
            const float val = kq[2 * DHH + d];
            const float bl  = kq[3 * DHH + d];
            const float beta = sigmoidf_(bl);
            const float vb = val * beta;
            const float db = (1.0f - beta) * tm;
            #pragma unroll
            for (int r = 0; r < RR; ++r)
                s_tv[r][d] = db * s_tv[r][d] + vb * occ[r];
        }
        __syncthreads();

        // waves 0..3: dot[r] = sum_j tk[r][j]*q[j] ; wave 4: den = sum_j s[j]*q[j]
        const int wv = tid >> 5, ln = tid & 31;
        if (wv < 5) {
            float part = 0.0f;
            if (wv < 4) {
                for (int j = ln; j < DKE_; j += 32) part += s_tk[wv][j] * s_q[j];
            } else {
                for (int j = ln; j < DKE_; j += 32) part += s_s[j] * s_q[j];
            }
            #pragma unroll
            for (int off = 16; off; off >>= 1) part += __shfl_down(part, off, 32);
            if (ln == 0) s_red[wv] = part;
        }
        __syncthreads();

        if (tid < DHH) {
            const float den = s_red[4] + 1e-6f;
            float num = 0.0f;
            #pragma unroll
            for (int r = 0; r < RR; ++r) num += s_red[r] * s_tv[r][tid];
            attn_bf[((size_t)t * BB + b) * (HH * DHH) + h * DHH + tid] = (bf16_t)(num / den);
        }
        __syncthreads();
    }
}

// ---------------------------------------------------------------------------
// Orchestration
// ---------------------------------------------------------------------------
extern "C" void kernel_launch(void* const* d_in, const int* in_sizes, int n_in,
                              void* d_out, int out_size, void* d_ws, size_t ws_size,
                              hipStream_t stream) {
    (void)in_sizes; (void)n_in; (void)out_size; (void)ws_size;

    const float* inputs = (const float*)d_in[0];
    const float* term   = (const float*)d_in[1];
    const float* tkp    = (const float*)d_in[2];
    const float* tvp    = (const float*)d_in[3];
    const float* sp     = (const float*)d_in[4];
    const float* tickp  = (const float*)d_in[5];
    const float* Wp     = (const float*)d_in[6];
    const float* bp     = (const float*)d_in[7];
    const float* Wa     = (const float*)d_in[8];
    const float* ba     = (const float*)d_in[9];
    const float* ln1g   = (const float*)d_in[10];
    const float* ln1b   = (const float*)d_in[11];
    const float* ln2g   = (const float*)d_in[12];
    const float* ln2b   = (const float*)d_in[13];
    const float* W1     = (const float*)d_in[14];
    const float* b1     = (const float*)d_in[15];
    const float* W2     = (const float*)d_in[16];
    const float* b2     = (const float*)d_in[17];
    const float* omg    = (const float*)d_in[18];

    const int MB = TT * BB; // 8192 rows

    char* ws = (char*)d_ws;
    size_t off = 0;
    auto take = [&](size_t bytes) -> char* {
        char* p = ws + off;
        off = (off + bytes + 255) & ~(size_t)255;
        return p;
    };

    float*  xF     = (float*) take((size_t)MB * DD * 4);        // LN1(x)  (residual)
    bf16_t* xH     = (bf16_t*)take((size_t)MB * DD * 2);        // LN1(x)  bf16
    float*  projF  = (float*) take((size_t)MB * NPROJ * 4);     // all_proj
    bf16_t* attnH  = (bf16_t*)take((size_t)MB * DD * 2);        // attn (T,B,H*DH)
    float*  x2F    = (float*) take((size_t)MB * DD * 4);        // x + attn_out
    bf16_t* ln2H   = (bf16_t*)take((size_t)MB * DD * 2);        // LN2(x2) bf16
    bf16_t* hH     = (bf16_t*)take((size_t)MB * DFFC_ * 2);     // relu(ffc hidden)
    bf16_t* WpH    = (bf16_t*)take((size_t)NPROJ * DD * 2);
    bf16_t* WaH    = (bf16_t*)take((size_t)DD * DD * 2);
    bf16_t* W1H    = (bf16_t*)take((size_t)DFFC_ * DD * 2);
    bf16_t* W2H    = (bf16_t*)take((size_t)DD * DFFC_ * 2);

    // 1) weight casts to bf16
    cast_f32_bf16<<<1024, 256, 0, stream>>>(Wp, WpH, NPROJ * DD);
    cast_f32_bf16<<<512,  256, 0, stream>>>(Wa, WaH, DD * DD);
    cast_f32_bf16<<<1024, 256, 0, stream>>>(W1, W1H, DFFC_ * DD);
    cast_f32_bf16<<<1024, 256, 0, stream>>>(W2, W2H, DD * DFFC_);

    // 2) LN1 -> xF (f32 residual) + xH (bf16 GEMM operand)
    ln_kernel<<<MB, 256, 0, stream>>>(inputs, ln1g, ln1b, xF, xH, DD);

    // 3) projection GEMM: all_proj = x @ Wp^T + bp   (8192 x 2608, K=512)
    {
        dim3 grid((NPROJ + 63) / 64, MB / 64);
        gemm_bf16_wmma<<<grid, 128, 0, stream>>>(xH, WpH, bp, nullptr,
                                                 projF, nullptr, MB, NPROJ, DD, 0);
    }

    // 4) gating + oscillator + discounted scans + attention readout
    scan_attn_kernel<<<BB * HH, 256, 0, stream>>>(projF, term, tkp, tvp, sp,
                                                  tickp, omg, attnH);

    // 5) x2 = x + attn @ Wa^T + ba   (8192 x 512, K=512)
    {
        dim3 grid(DD / 64, MB / 64);
        gemm_bf16_wmma<<<grid, 128, 0, stream>>>(attnH, WaH, ba, xF,
                                                 x2F, nullptr, MB, DD, DD, 0);
    }

    // 6) LN2 -> bf16
    ln_kernel<<<MB, 256, 0, stream>>>(x2F, ln2g, ln2b, nullptr, ln2H, DD);

    // 7) hidden = relu(ln2 @ W1^T + b1)   (8192 x 2048, K=512)
    {
        dim3 grid(DFFC_ / 64, MB / 64);
        gemm_bf16_wmma<<<grid, 128, 0, stream>>>(ln2H, W1H, b1, nullptr,
                                                 nullptr, hH, MB, DFFC_, DD, 1);
    }

    // 8) out = x2 + hidden @ W2^T + b2    (8192 x 512, K=2048)
    {
        dim3 grid(DD / 64, MB / 64);
        gemm_bf16_wmma<<<grid, 128, 0, stream>>>(hH, W2H, b2, x2F,
                                                 (float*)d_out, nullptr, MB, DD, DFFC_, 0);
    }
}
